// MDNLayer_32195074851216
// MI455X (gfx1250) — matrix operand order repaired and checked by reading, not compiled, not run on hardware
//
#include <hip/hip_runtime.h>
#include <hip/hip_bf16.h>
#include <math.h>

// ---------------------------------------------------------------------------
// MDN edge-MLP, fully fused, bf16 WMMA (v_wmma_f32_16x16x32_bf16), fp32 accum.
// ---------------------------------------------------------------------------

#define N_EDGES   500000
#define D_IN      128
#define HID       512
#define G_MIX     10
#define BN_EPS    1e-5f

#define TILE_M    64            // edges per workgroup
#define THREADS   256           // 8 waves (wave32)
#define XS        520           // LDS row stride (bf16) for 512-wide activations (pad 8)
#define CS        264           // LDS row stride (bf16) for 256-wide concat input (pad 8)
#define SMEM_X_BYTES   (TILE_M * XS * 2)          // 66560
#define SMEM_HEAD_OFF  SMEM_X_BYTES
#define SMEM_BYTES     (SMEM_X_BYTES + TILE_M * 32 * 4)  // + 8KB head buffer = 74752

typedef __attribute__((ext_vector_type(16))) __bf16 v16bf;
typedef __attribute__((ext_vector_type(8)))  __bf16 v8bf;
typedef __attribute__((ext_vector_type(8)))  float  v8f;

// ---- ws layout (bf16 elements) --------------------------------------------
// W0:    256x512 fragment-major            : 131072 elems
// Wh[2]: 512x512 fragment-major, each      : 262144 elems
// Heads: 512x32  fragment-major (pi|sig|mu): 16384  elems
#define WS_W0_OFF  0
#define WS_WH_OFF  131072
#define WS_HD_OFF  (131072 + 2 * 262144)

// ---------------------------------------------------------------------------
// Weight preprocessing: row-major fp32 KxN  ->  bf16 fragment-major layout.
// Fragment (kt,nt) is a 32x16 B-tile stored as [lane][16 contiguous K].
//   lane: n = nt*16 + (lane&15), K-half = (lane>>4)*16
//   elem i (0..15): k = kt*32 + Khalf + i
// ---------------------------------------------------------------------------
__global__ void prep_b_kernel(const float* __restrict__ W, __bf16* __restrict__ dst,
                              int K, int N) {
    int elem = blockIdx.x * blockDim.x + threadIdx.x;
    int total = K * N;
    if (elem >= total) return;
    int i    = elem & 15;
    int lane = (elem >> 4) & 31;
    int tk   = elem >> 9;
    int KT   = K >> 5;
    int kt   = tk % KT;
    int nt   = tk / KT;
    int k = kt * 32 + ((lane >> 4) << 4) + i;
    int n = nt * 16 + (lane & 15);
    dst[elem] = (__bf16)W[(size_t)k * N + n];
}

// Heads: build padded 512x32 matrix [W_pi | W_sigma | W_mu | 0 0] on the fly.
__global__ void prep_heads_kernel(const float* __restrict__ Wpi,
                                  const float* __restrict__ Wsig,
                                  const float* __restrict__ Wmu,
                                  __bf16* __restrict__ dst) {
    int elem = blockIdx.x * blockDim.x + threadIdx.x;
    const int K = HID, N = 32, KT = K >> 5;
    if (elem >= K * N) return;
    int i    = elem & 15;
    int lane = (elem >> 4) & 31;
    int tk   = elem >> 9;
    int kt   = tk % KT;
    int nt   = tk / KT;
    int k = kt * 32 + ((lane >> 4) << 4) + i;
    int n = nt * 16 + (lane & 15);
    float v = 0.0f;
    if      (n < 10) v = Wpi [(size_t)k * G_MIX + n];
    else if (n < 20) v = Wsig[(size_t)k * G_MIX + (n - 10)];
    else if (n < 30) v = Wmu [(size_t)k * G_MIX + (n - 20)];
    dst[elem] = (__bf16)v;
}

// ---------------------------------------------------------------------------
// A-fragment load from LDS: 16x32 bf16 tile in contiguous k order (same order as the B prep).
//   lane: m = lane&15, K-half h = lane>>4
//   elem i: k = 16h + i   ->  one contiguous 16-elem run.
// ---------------------------------------------------------------------------
__device__ __forceinline__ v16bf load_a_frag(const __bf16* __restrict__ Xs,
                                             int stride, int mbase, int kbase,
                                             int lane) {
    int m = lane & 15;
    int h = lane >> 4;
    const __bf16* p = Xs + (mbase + m) * stride + kbase + 16 * h;
    v8bf lo = *(const v8bf*)(p);
    v8bf hi = *(const v8bf*)(p + 8);
    return __builtin_shufflevector(lo, hi, 0,1,2,3,4,5,6,7,8,9,10,11,12,13,14,15);
}

__device__ __forceinline__ float elu_f(float x) {
    return x > 0.0f ? x : (expf(x) - 1.0f);
}

// ---------------------------------------------------------------------------
// Fused main kernel: gather -> (GEMM+BN+ELU) x3 -> heads -> softmax/elu -> out
// ---------------------------------------------------------------------------
__global__ __launch_bounds__(THREADS)
void mdn_fused_kernel(const float* __restrict__ h,
                      const int*   __restrict__ pl_edge,
                      const float* __restrict__ pl_dist,
                      const float* __restrict__ b0,
                      const float* __restrict__ bh,
                      const float* __restrict__ bn_gamma,
                      const float* __restrict__ bn_beta,
                      const float* __restrict__ bn_mean,
                      const float* __restrict__ bn_var,
                      const float* __restrict__ b_pi,
                      const float* __restrict__ b_sigma,
                      const float* __restrict__ b_mu,
                      const __bf16* __restrict__ ws,
                      float* __restrict__ out) {
    extern __shared__ char smem[];
    __bf16* Xs = (__bf16*)smem;                      // activations / concat input
    float*  Hs = (float*)(smem + SMEM_HEAD_OFF);     // 64 x 32 head staging

    const int tid  = threadIdx.x;
    const int lane = tid & 31;
    const int wid  = tid >> 5;
    const int ebase = blockIdx.x * TILE_M;

    const __bf16* wsW0 = ws + WS_W0_OFF;
    const __bf16* wsWh = ws + WS_WH_OFF;
    const __bf16* wsHd = ws + WS_HD_OFF;

    // ------------------ Phase 1: gather + concat -> LDS (bf16) -------------
    {
        int e_loc = tid >> 2;          // 64 rows
        int part  = tid & 3;           // 4 x 64 floats per row
        int e = ebase + e_loc;
        int idx1 = (e < N_EDGES) ? pl_edge[N_EDGES + e] : 0;  // h_l (first 128)
        int idx0 = (e < N_EDGES) ? pl_edge[e]           : 0;  // h_p (next 128)
        const float* src = (part < 2)
            ? h + (size_t)idx1 * D_IN + part * 64
            : h + (size_t)idx0 * D_IN + (part - 2) * 64;
        __bf16* dstrow = Xs + e_loc * CS + part * 64;
        #pragma unroll
        for (int j = 0; j < 64; j += 4) {
            float4 v = *(const float4*)(src + j);
            dstrow[j + 0] = (__bf16)v.x;
            dstrow[j + 1] = (__bf16)v.y;
            dstrow[j + 2] = (__bf16)v.z;
            dstrow[j + 3] = (__bf16)v.w;
        }
    }
    __syncthreads();

    const int nbase = wid * 64;        // this wave's 64-column slab

    // ------------------ Phase 2: layer 0 (K=256) ---------------------------
    {
        v8f acc[4][4];
        const v8f vz = {0.f,0.f,0.f,0.f,0.f,0.f,0.f,0.f};
        #pragma unroll
        for (int mt = 0; mt < 4; ++mt)
            #pragma unroll
            for (int nt = 0; nt < 4; ++nt) acc[mt][nt] = vz;

        #pragma unroll
        for (int kt = 0; kt < 8; ++kt) {
            v16bf aF[4], bF[4];
            #pragma unroll
            for (int mt = 0; mt < 4; ++mt)
                aF[mt] = load_a_frag(Xs, CS, mt * 16, kt * 32, lane);
            #pragma unroll
            for (int nt = 0; nt < 4; ++nt) {
                int ntG = (nbase >> 4) + nt;
                bF[nt] = *(const v16bf*)(wsW0 + ((size_t)(ntG * 8 + kt) * 32 + lane) * 16);
            }
            #pragma unroll
            for (int mt = 0; mt < 4; ++mt)
                #pragma unroll
                for (int nt = 0; nt < 4; ++nt)
                    acc[mt][nt] = __builtin_amdgcn_wmma_f32_16x16x32_bf16(
                        false, aF[mt], false, bF[nt], (short)0, acc[mt][nt],
                        false, false);
        }
        __syncthreads();   // all reads of concat-input done before overwrite

        // BN(layer0) + ELU epilogue, write bf16 activations into Xs
        #pragma unroll
        for (int nt = 0; nt < 4; ++nt) {
            int col = nbase + nt * 16 + (lane & 15);
            float g  = bn_gamma[col], be = bn_beta[col];
            float mn = bn_mean[col],  vr = bn_var[col];
            float sA = g * rsqrtf(vr + BN_EPS);
            float sB = sA * (b0[col] - mn) + be;
            #pragma unroll
            for (int mt = 0; mt < 4; ++mt) {
                #pragma unroll
                for (int r = 0; r < 8; ++r) {
                    int row = mt * 16 + r + 8 * (lane >> 4);
                    float y = elu_f(sA * acc[mt][nt][r] + sB);
                    Xs[row * XS + col] = (__bf16)y;
                }
            }
        }
    }
    __syncthreads();

    // ------------------ Phase 3: hidden layers (K=512) ---------------------
    for (int L = 0; L < 2; ++L) {
        const __bf16* wB = wsWh + (size_t)L * 262144;
        v8f acc[4][4];
        const v8f vz = {0.f,0.f,0.f,0.f,0.f,0.f,0.f,0.f};
        #pragma unroll
        for (int mt = 0; mt < 4; ++mt)
            #pragma unroll
            for (int nt = 0; nt < 4; ++nt) acc[mt][nt] = vz;

        for (int kt = 0; kt < 16; ++kt) {
            v16bf aF[4], bF[4];
            #pragma unroll
            for (int mt = 0; mt < 4; ++mt)
                aF[mt] = load_a_frag(Xs, XS, mt * 16, kt * 32, lane);
            #pragma unroll
            for (int nt = 0; nt < 4; ++nt) {
                int ntG = (nbase >> 4) + nt;
                bF[nt] = *(const v16bf*)(wB + ((size_t)(ntG * 16 + kt) * 32 + lane) * 16);
            }
            #pragma unroll
            for (int mt = 0; mt < 4; ++mt)
                #pragma unroll
                for (int nt = 0; nt < 4; ++nt)
                    acc[mt][nt] = __builtin_amdgcn_wmma_f32_16x16x32_bf16(
                        false, aF[mt], false, bF[nt], (short)0, acc[mt][nt],
                        false, false);
        }
        __syncthreads();   // all reads of Xs done before in-place update

        const float* bias = bh + (size_t)L * HID;
        const float* gp = bn_gamma + (size_t)(L + 1) * HID;
        const float* bp = bn_beta  + (size_t)(L + 1) * HID;
        const float* mp = bn_mean  + (size_t)(L + 1) * HID;
        const float* vp = bn_var   + (size_t)(L + 1) * HID;
        #pragma unroll
        for (int nt = 0; nt < 4; ++nt) {
            int col = nbase + nt * 16 + (lane & 15);
            float sA = gp[col] * rsqrtf(vp[col] + BN_EPS);
            float sB = sA * (bias[col] - mp[col]) + bp[col];
            #pragma unroll
            for (int mt = 0; mt < 4; ++mt) {
                #pragma unroll
                for (int r = 0; r < 8; ++r) {
                    int row = mt * 16 + r + 8 * (lane >> 4);
                    float y = elu_f(sA * acc[mt][nt][r] + sB);
                    Xs[row * XS + col] = (__bf16)y;
                }
            }
        }
        __syncthreads();
    }

    // ------------------ Phase 4: heads (K=512, N=32 padded) ----------------
    {
        int mt  = wid & 3;     // 4 M-tiles
        int nt2 = wid >> 2;    // 2 N-tiles -> exactly 8 waves
        v8f acc = {0.f,0.f,0.f,0.f,0.f,0.f,0.f,0.f};
        for (int kt = 0; kt < 16; ++kt) {
            v16bf aF = load_a_frag(Xs, XS, mt * 16, kt * 32, lane);
            v16bf bF = *(const v16bf*)(wsHd + ((size_t)(nt2 * 16 + kt) * 32 + lane) * 16);
            acc = __builtin_amdgcn_wmma_f32_16x16x32_bf16(
                false, aF, false, bF, (short)0, acc, false, false);
        }
        int col = nt2 * 16 + (lane & 15);
        #pragma unroll
        for (int r = 0; r < 8; ++r) {
            int row = mt * 16 + r + 8 * (lane >> 4);
            float x = acc[r];
            float val;
            if      (col < 10) val = x + b_pi[col];                         // raw logits
            else if (col < 20) val = elu_f(x + b_sigma[col - 10]) + 1.1f;   // sigma
            else if (col < 30) val = elu_f(x + b_mu[col - 20]) + 1.0f;      // mu
            else               val = 0.0f;
            Hs[row * 32 + col] = val;
        }
    }
    __syncthreads();

    // ------------------ Phase 5: softmax + output writes -------------------
    if (tid < TILE_M) {
        int e = ebase + tid;
        if (e < N_EDGES) {
            const float* hr = Hs + tid * 32;
            float m = hr[0];
            #pragma unroll
            for (int g = 1; g < G_MIX; ++g) m = fmaxf(m, hr[g]);
            float s = 0.0f, ex[G_MIX];
            #pragma unroll
            for (int g = 0; g < G_MIX; ++g) { ex[g] = expf(hr[g] - m); s += ex[g]; }
            float inv = 1.0f / s;
            float* out_pi  = out;
            float* out_sig = out + (size_t)N_EDGES * G_MIX;
            float* out_mu  = out + (size_t)2 * N_EDGES * G_MIX;
            float* out_dst = out + (size_t)3 * N_EDGES * G_MIX;
            #pragma unroll
            for (int g = 0; g < G_MIX; ++g) {
                out_pi [(size_t)e * G_MIX + g] = ex[g] * inv;
                out_sig[(size_t)e * G_MIX + g] = hr[10 + g];
                out_mu [(size_t)e * G_MIX + g] = hr[20 + g];
            }
            out_dst[e] = pl_dist[e];
        }
    }
}

// ---------------------------------------------------------------------------
extern "C" void kernel_launch(void* const* d_in, const int* in_sizes, int n_in,
                              void* d_out, int out_size, void* d_ws, size_t ws_size,
                              hipStream_t stream) {
    const float* h        = (const float*)d_in[0];
    const int*   pl_edge  = (const int*)  d_in[1];
    const float* pl_dist  = (const float*)d_in[2];
    const float* W0       = (const float*)d_in[3];
    const float* b0       = (const float*)d_in[4];
    const float* Wh       = (const float*)d_in[5];
    const float* bh       = (const float*)d_in[6];
    const float* bn_gamma = (const float*)d_in[7];
    const float* bn_beta  = (const float*)d_in[8];
    const float* bn_mean  = (const float*)d_in[9];
    const float* bn_var   = (const float*)d_in[10];
    const float* W_pi     = (const float*)d_in[11];
    const float* b_pi     = (const float*)d_in[12];
    const float* W_sigma  = (const float*)d_in[13];
    const float* b_sigma  = (const float*)d_in[14];
    const float* W_mu     = (const float*)d_in[15];
    const float* b_mu     = (const float*)d_in[16];
    float* out = (float*)d_out;

    __bf16* ws = (__bf16*)d_ws;

    // Weight conversion to bf16 fragment-major (re-done each call: deterministic)
    prep_b_kernel<<<(131072 + 255) / 256, 256, 0, stream>>>(W0, ws + WS_W0_OFF, 256, 512);
    prep_b_kernel<<<(262144 + 255) / 256, 256, 0, stream>>>(Wh, ws + WS_WH_OFF, 512, 512);
    prep_b_kernel<<<(262144 + 255) / 256, 256, 0, stream>>>(Wh + 262144,
                                                            ws + WS_WH_OFF + 262144, 512, 512);
    prep_heads_kernel<<<(16384 + 255) / 256, 256, 0, stream>>>(W_pi, W_sigma, W_mu,
                                                               ws + WS_HD_OFF);

    int grid = (N_EDGES + TILE_M - 1) / TILE_M;   // 7813
    mdn_fused_kernel<<<grid, THREADS, SMEM_BYTES, stream>>>(
        h, pl_edge, pl_dist, b0, bh, bn_gamma, bn_beta, bn_mean, bn_var,
        b_pi, b_sigma, b_mu, ws, out);
}
